// Qnet_16037407883355
// MI455X (gfx1250) — matrix operand order repaired and checked by reading, not compiled, not run on hardware
//
#include <hip/hip_runtime.h>
#include <math.h>

#define N_NODES   50000
#define N_EDGES   800000
#define HEADS     4
#define OUT_CH    32
#define CH        128        // HEADS*OUT_CH
#define NEG_SLOPE 0.2f

typedef __attribute__((ext_vector_type(16))) __bf16 v16bf;
typedef __attribute__((ext_vector_type(8)))  float  v8f;

static __device__ __forceinline__ __bf16 f2bf(float x) { return (__bf16)x; }

// float atomic max via monotone int mapping (init must be -inf)
static __device__ __forceinline__ void atomicMaxF(float* addr, float value) {
    if (value >= 0.f) atomicMax((int*)addr, __float_as_int(value));
    else              atomicMin((unsigned int*)addr, __float_as_uint(value));
}

// ---------------------------------------------------------------------------
// Pack W (128x128 f32, row-major) into bf16 B-fragments for wmma 16x16x32,
// and fold We (8x16) with a_e (4x4) into ce[h][8] so the edge-embedding GEMM
// disappears:  s_e[e,h] = dot8(edge_attr[e], ce[h]).
// Bp index: ((ntile*4 + kstep)*32 + lane)*16 ; lanes 0-15: K=kk..kk+7,kk+16..kk+23,
// lanes 16-31: K=kk+8..kk+15,kk+24..kk+31, N = ntile*16 + lane%16.
// ---------------------------------------------------------------------------
__global__ void pack_b_kernel(const float* __restrict__ W,
                              const float* __restrict__ We,
                              const float* __restrict__ att,
                              __bf16* __restrict__ Bp,
                              float* __restrict__ ce) {
    int t = blockIdx.x * blockDim.x + threadIdx.x;   // 0..1023
    if (t < 1024) {
        int lane = t & 31;
        int kk   = (t >> 5) & 3;        // K step (0..3) -> K base kk*32
        int nt   = t >> 7;              // N tile (0..7)
        int n    = nt * 16 + (lane & 15);
        int kb   = (lane >> 4) * 8;
        __bf16* dst = Bp + (size_t)t * 16;
        #pragma unroll
        for (int i = 0; i < 16; ++i) {
            int k = kk * 32 + kb + (i & 7) + ((i >> 3) << 4);
            dst[i] = f2bf(W[k * CH + n]);
        }
    }
    if (blockIdx.x == 0 && threadIdx.x < 32) {
        int h = threadIdx.x >> 3, r = threadIdx.x & 7;
        float acc = 0.f;
        #pragma unroll
        for (int p = 0; p < 4; ++p)
            acc += We[r * 16 + h * 4 + p] * att[h * 68 + 64 + p];
        ce[h * 8 + r] = acc;
    }
}

// ---------------------------------------------------------------------------
// out = bias (aggregation accumulates on top); m = -inf; denom = 0
// ---------------------------------------------------------------------------
__global__ void init_kernel(float* __restrict__ out, const float* __restrict__ bias,
                            float* __restrict__ m, float* __restrict__ denom) {
    int i = blockIdx.x * blockDim.x + threadIdx.x;
    if (i < N_NODES * CH)    out[i] = bias[i & (CH - 1)];
    if (i < N_NODES * HEADS) { m[i] = -INFINITY; denom[i] = 0.f; }
}

// ---------------------------------------------------------------------------
// xh = x @ W  via v_wmma_f32_16x16x32_bf16.
// One wave per 16x16 output tile: block = 8 waves = the 8 N-tiles,
// grid.x = 3125 M-tiles (50000 = 3125*16 exactly -> no bounds check,
// EXEC stays all-ones as WMMA requires).
// ---------------------------------------------------------------------------
__global__ void gemm_xh_wmma(const float* __restrict__ x,
                             const __bf16* __restrict__ Bp,
                             float* __restrict__ xh) {
    const int lane = threadIdx.x & 31;
    const int wave = threadIdx.x >> 5;          // N tile 0..7
    const int m0   = blockIdx.x * 16;
    const int row  = m0 + (lane & 15);
    const int kb   = (lane >> 4) * 8;           // A-layout K sub-base
    const float* xr = x + (size_t)row * CH;

    v8f c = {};
    #pragma unroll
    for (int kk = 0; kk < 4; ++kk) {
        const float* abase = xr + kk * 32 + kb;
        v16bf a;
        #pragma unroll
        for (int i = 0; i < 8; ++i) a[i]     = f2bf(abase[i]);        // K = kb+0..7
        #pragma unroll
        for (int i = 0; i < 8; ++i) a[8 + i] = f2bf(abase[16 + i]);   // K = kb+16..23
        v16bf b = *(const v16bf*)(Bp + (size_t)((wave * 4 + kk) * 32 + lane) * 16);
        c = __builtin_amdgcn_wmma_f32_16x16x32_bf16(
                false, a, false, b, (short)0, c, false, false);
    }

    const int n     = wave * 16 + (lane & 15);
    const int mbase = m0 + (lane >> 4) * 8;
    #pragma unroll
    for (int r = 0; r < 8; ++r)
        xh[(size_t)(mbase + r) * CH + n] = c[r];
}

// ---------------------------------------------------------------------------
// Per (node, head): s_i = xh[n,h,:]·a_i[h], s_j = xh[n,h,:]·a_j[h]
// ---------------------------------------------------------------------------
__global__ void node_scores(const float* __restrict__ xh, const float* __restrict__ att,
                            float* __restrict__ si, float* __restrict__ sj) {
    int t = blockIdx.x * blockDim.x + threadIdx.x;
    if (t >= N_NODES * HEADS) return;
    int h = t & 3, n = t >> 2;
    const float* v  = xh + (size_t)n * CH + h * OUT_CH;
    const float* ai = att + h * 68;
    const float* aj = ai + OUT_CH;
    float accI = 0.f, accJ = 0.f;
    #pragma unroll
    for (int o = 0; o < OUT_CH; ++o) { float xv = v[o]; accI += xv * ai[o]; accJ += xv * aj[o]; }
    si[t] = accI; sj[t] = accJ;
}

// ---------------------------------------------------------------------------
// Pass 1 (edges): alpha = lrelu(s_i[dst] + s_j[src] + s_e); segment max into m
// ---------------------------------------------------------------------------
__global__ void edge_alpha(const int* __restrict__ ei, const float* __restrict__ eattr,
                           const float* __restrict__ ce, const float* __restrict__ si,
                           const float* __restrict__ sj, float* __restrict__ alpha,
                           float* __restrict__ m) {
    int e = blockIdx.x * blockDim.x + threadIdx.x;
    if (e >= N_EDGES) return;
    int src = ei[e], dst = ei[N_EDGES + e];
    float ea8[8];
    #pragma unroll
    for (int i = 0; i < 8; ++i) ea8[i] = eattr[(size_t)e * 8 + i];
    #pragma unroll
    for (int h = 0; h < HEADS; ++h) {
        float se = 0.f;
        #pragma unroll
        for (int i = 0; i < 8; ++i) se += ea8[i] * ce[h * 8 + i];
        float a = si[dst * 4 + h] + sj[src * 4 + h] + se;
        a = a > 0.f ? a : NEG_SLOPE * a;
        alpha[(size_t)e * 4 + h] = a;
        atomicMaxF(&m[dst * 4 + h], a);
    }
}

// ---------------------------------------------------------------------------
// Pass 2 (edges): ea = exp(alpha - m[dst]) (stored in place); denom += ea
// ---------------------------------------------------------------------------
__global__ void edge_exp(const int* __restrict__ ei, float* __restrict__ alpha,
                         const float* __restrict__ m, float* __restrict__ denom) {
    int e = blockIdx.x * blockDim.x + threadIdx.x;
    if (e >= N_EDGES) return;
    int dst = ei[N_EDGES + e];
    #pragma unroll
    for (int h = 0; h < HEADS; ++h) {
        float v = __expf(alpha[(size_t)e * 4 + h] - m[dst * 4 + h]);
        alpha[(size_t)e * 4 + h] = v;
        atomicAdd(&denom[dst * 4 + h], v);
    }
}

// ---------------------------------------------------------------------------
// Pass 3: one wave per edge. Lane l handles channels l, l+32, l+64, l+96
// (one per head) -> consecutive lanes hit consecutive addresses: coalesced
// L2-resident gathers of xh[src] and cacheline-dense atomic adds into out[dst].
// ---------------------------------------------------------------------------
__global__ void edge_scatter(const int* __restrict__ ei, const float* __restrict__ alpha,
                             const float* __restrict__ denom, const float* __restrict__ xh,
                             float* __restrict__ out) {
    int wave = threadIdx.x >> 5;
    int lane = threadIdx.x & 31;
    int e = blockIdx.x * 8 + wave;
    if (e >= N_EDGES) return;
    int src = ei[e], dst = ei[N_EDGES + e];
    float4 ev = *(const float4*)(alpha + (size_t)e * 4);
    float4 dv = *(const float4*)(denom + (size_t)dst * 4);
    float c0 = ev.x / (dv.x + 1e-16f);
    float c1 = ev.y / (dv.y + 1e-16f);
    float c2 = ev.z / (dv.z + 1e-16f);
    float c3 = ev.w / (dv.w + 1e-16f);
    const float* xs = xh + (size_t)src * CH;
    float* od = out + (size_t)dst * CH;
    atomicAdd(&od[lane],      xs[lane]      * c0);
    atomicAdd(&od[32 + lane], xs[32 + lane] * c1);
    atomicAdd(&od[64 + lane], xs[64 + lane] * c2);
    atomicAdd(&od[96 + lane], xs[96 + lane] * c3);
}

// ---------------------------------------------------------------------------
extern "C" void kernel_launch(void* const* d_in, const int* in_sizes, int n_in,
                              void* d_out, int out_size, void* d_ws, size_t ws_size,
                              hipStream_t stream) {
    (void)in_sizes; (void)n_in; (void)out_size; (void)ws_size;
    const float* x     = (const float*)d_in[0];
    const int*   ei    = (const int*)  d_in[1];
    const float* eattr = (const float*)d_in[2];
    const float* W     = (const float*)d_in[3];
    const float* We    = (const float*)d_in[4];
    const float* att   = (const float*)d_in[5];
    const float* bias  = (const float*)d_in[6];
    float* out = (float*)d_out;

    // workspace layout (floats)
    float* ws    = (float*)d_ws;
    float* xh    = ws;                       // 6,400,000
    float* si    = xh + (size_t)N_NODES * CH;        // 200,000
    float* sj    = si + (size_t)N_NODES * HEADS;     // 200,000
    float* m     = sj + (size_t)N_NODES * HEADS;     // 200,000
    float* denom = m  + (size_t)N_NODES * HEADS;     // 200,000
    float* alpha = denom + (size_t)N_NODES * HEADS;  // 3,200,000
    float* ce    = alpha + (size_t)N_EDGES * HEADS;  // 32
    __bf16* Bp   = (__bf16*)(ce + 32);               // 16,384 bf16 (32-byte aligned)

    pack_b_kernel<<<4, 256, 0, stream>>>(W, We, att, Bp, ce);
    init_kernel<<<(N_NODES * CH + 255) / 256, 256, 0, stream>>>(out, bias, m, denom);
    gemm_xh_wmma<<<N_NODES / 16, 256, 0, stream>>>(x, Bp, xh);
    node_scores<<<(N_NODES * HEADS + 255) / 256, 256, 0, stream>>>(xh, att, si, sj);
    edge_alpha<<<N_EDGES / 256, 256, 0, stream>>>(ei, eattr, ce, si, sj, alpha, m);
    edge_exp<<<N_EDGES / 256, 256, 0, stream>>>(ei, alpha, m, denom);
    edge_scatter<<<(N_EDGES + 7) / 8, 256, 0, stream>>>(ei, alpha, denom, xh, out);
}
